// CLAM_SB_14499809591303
// MI455X (gfx1250) — matrix operand order repaired and checked by reading, not compiled, not run on hardware
//
#include <hip/hip_runtime.h>
#include <hip/hip_bf16.h>
#include <cmath>

// ---------------------------------------------------------------------------
// CLAM-SB fused forward for MI455X (gfx1250), wave32 + bf16 WMMA.
// ---------------------------------------------------------------------------

typedef __bf16 bf16_t;
typedef __attribute__((ext_vector_type(16))) __bf16 v16bf;
typedef __attribute__((ext_vector_type(8)))  __bf16 v8bf;
typedef __attribute__((ext_vector_type(8)))  float   v8f;

#define NROWS   100000
#define DIN     1024
#define HDIM    512
#define DDIM    256
#define KSAMP   8

#define TROWS   64          // rows per block in fused kernel
#define XSTR    528         // LDS x-tile row stride (bf16 elems), 16B-friendly pad

// Pack two f32 -> two bf16 in one u32 (low = a, high = b).
__device__ __forceinline__ unsigned pack_bf16x2(float a, float b) {
  union { bf16_t h[2]; unsigned u; } u_;
  u_.h[0] = (bf16_t)a;
  u_.h[1] = (bf16_t)b;
  return u_.u;
}

// Build a 16-element bf16 fragment from two contiguous 8-element (16B) runs.
__device__ __forceinline__ v16bf frag_pair(const bf16_t* p0, const bf16_t* p1) {
  v8bf lo = *(const v8bf*)p0;
  v8bf hi = *(const v8bf*)p1;
  v16bf r;
#pragma unroll
  for (int i = 0; i < 8; ++i) { r[i] = lo[i]; r[i + 8] = hi[i]; }
  return r;
}

__device__ __forceinline__ v8f wmma_bf16(v16bf a, v16bf b, v8f c) {
  // D = A(16x32 bf16) * B(32x16 bf16) + C(16x16 f32)
  return __builtin_amdgcn_wmma_f32_16x16x32_bf16(false, a, false, b, (short)0, c,
                                                 false, false);
}

// ---------------------------------------------------------------------------
// Kernel 1: fused  x = relu(h@W1+b1);  s = ((tanh(x@Wa+ba)*sigmoid(x@Wb+bb))@Wc+bc)
// Writes: xbf (bf16 [N][512]), A_raw (d_out+5), two mutable score copies.
// Block: 256 threads (8 wave32), 64 rows.
// ---------------------------------------------------------------------------
__global__ void __launch_bounds__(256)
k_fused(const float* __restrict__ h,  const float* __restrict__ W1,
        const float* __restrict__ b1, const float* __restrict__ Wa,
        const float* __restrict__ ba, const float* __restrict__ Wb,
        const float* __restrict__ bb, const float* __restrict__ Wc,
        const float* __restrict__ bc,
        bf16_t* __restrict__ xbf, float* __restrict__ a_raw,
        float* __restrict__ scoresA, float* __restrict__ scoresB) {
  __shared__ bf16_t lds_x[TROWS * XSTR];   // 67584 B : x tile, bf16
  __shared__ bf16_t lds_w[HDIM * 32];      // 32768 B : W chunk(s), transposed [n][k]
  __shared__ bf16_t lds_h[TROWS * 32];     //  4096 B : A-side chunk
  __shared__ float  lds_s[TROWS];          // per-row attention score accum

  unsigned* lds_h32 = (unsigned*)lds_h;    // [64][16] packed k-pairs
  unsigned* lds_w32 = (unsigned*)lds_w;    // [n][16]  packed k-pairs

  const int t    = threadIdx.x;
  const int lane = t & 31;
  const int wave = t >> 5;
  const int half = lane >> 4;   // 0/1 : lane group
  const int ln   = lane & 15;   // column-in-tile
  const int row0 = blockIdx.x * TROWS;

  // ----------------- Phase 1: x = relu(h @ W1 + b1) ------------------------
  const int mtile = wave >> 1;          // 0..3  -> rows [mtile*16, +16)
  const int nbase = (wave & 1) * 16;    // 16 n-tiles per wave -> cols 0..511

  v8f acc[16] = {};

  for (int kc = 0; kc < DIN / 32; ++kc) {
    // stage h chunk [64 rows][32 k] -> bf16 row-major; pack pairs -> b64 store
#pragma unroll
    for (int i = 0; i < 2; ++i) {
      int id = t + 256 * i;                  // 0..511 float4s
      int r  = id >> 3;
      int c4 = (id & 7) * 4;                 // k within chunk
      int gr = row0 + r; if (gr >= NROWS) gr = NROWS - 1;
      float4 v = *(const float4*)(h + (size_t)gr * DIN + kc * 32 + c4);
      uint2 p;
      p.x = pack_bf16x2(v.x, v.y);
      p.y = pack_bf16x2(v.z, v.w);
      *(uint2*)&lds_h32[r * 16 + (c4 >> 1)] = p;
    }
    // stage W1 chunk [32 k][512 n] -> transposed lds_w[n][k]; pack (k,k+1)
#pragma unroll
    for (int i = 0; i < 8; ++i) {
      int id  = t + 256 * i;                 // 0..2047 tasks
      int kk2 = id >> 7;                     // 0..15 : k-pair index
      int c4  = (id & 127) * 4;              // n
      const float* g0 = W1 + (size_t)(kc * 32 + 2 * kk2 + 0) * HDIM + c4;
      const float* g1 = W1 + (size_t)(kc * 32 + 2 * kk2 + 1) * HDIM + c4;
      float4 v0 = *(const float4*)g0;
      float4 v1 = *(const float4*)g1;
      lds_w32[(c4 + 0) * 16 + kk2] = pack_bf16x2(v0.x, v1.x);
      lds_w32[(c4 + 1) * 16 + kk2] = pack_bf16x2(v0.y, v1.y);
      lds_w32[(c4 + 2) * 16 + kk2] = pack_bf16x2(v0.z, v1.z);
      lds_w32[(c4 + 3) * 16 + kk2] = pack_bf16x2(v0.w, v1.w);
    }
    __syncthreads();

    // A frag: row m, K run [16h..16h+15]
    const bf16_t* arow = &lds_h[(mtile * 16 + ln) * 32];
    v16bf af = frag_pair(arow + 16 * half, arow + 16 * half + 8);
#pragma unroll
    for (int ntl = 0; ntl < 16; ++ntl) {
      const bf16_t* brow = &lds_w[((nbase + ntl) * 16 + ln) * 32 + 16 * half];
      v16bf bf = frag_pair(brow, brow + 8);
      acc[ntl] = wmma_bf16(af, bf, acc[ntl]);
    }
    __syncthreads();
  }

  // bias + relu -> lds_x (bf16)
#pragma unroll
  for (int ntl = 0; ntl < 16; ++ntl) {
    int n = (nbase + ntl) * 16 + ln;
    float bias = b1[n];
#pragma unroll
    for (int r = 0; r < 8; ++r) {
      int m = mtile * 16 + half * 8 + r;
      float v = acc[ntl][r] + bias;
      v = v > 0.f ? v : 0.f;
      lds_x[m * XSTR + n] = (bf16_t)v;
    }
  }
  if (t < TROWS) lds_s[t] = 0.f;
  __syncthreads();

  // stream x tile to global workspace (bf16, coalesced b128)
#pragma unroll
  for (int i = 0; i < 16; ++i) {
    int id = t + 256 * i;                    // 0..4095 8-elem groups
    int r  = id >> 6;
    int c8 = (id & 63) * 8;
    int gr = row0 + r;
    if (gr < NROWS) {
      v8bf v = *(const v8bf*)&lds_x[r * XSTR + c8];
      *(v8bf*)(xbf + (size_t)gr * HDIM + c8) = v;
    }
  }
  __syncthreads();

  // ----------------- Phase 2: gated attention score ------------------------
  const int nb2 = (wave & 1) * 8;           // 8 d-tiles per wave -> d 0..255
  bf16_t*   lds_wa   = lds_w;               // [256][32]
  bf16_t*   lds_wb   = lds_w + DDIM * 32;   // [256][32]
  unsigned* lds_wa32 = (unsigned*)lds_wa;
  unsigned* lds_wb32 = (unsigned*)lds_wb;

  v8f acca[8] = {};
  v8f accg[8] = {};

  for (int kc = 0; kc < HDIM / 32; ++kc) {
#pragma unroll
    for (int i = 0; i < 4; ++i) {
      int id  = t + 256 * i;                 // 0..1023 tasks
      int kk2 = id >> 6;                     // 0..15 : k-pair index
      int c4  = (id & 63) * 4;               // d
      size_t o0 = (size_t)(kc * 32 + 2 * kk2 + 0) * DDIM + c4;
      size_t o1 = (size_t)(kc * 32 + 2 * kk2 + 1) * DDIM + c4;
      float4 a0 = *(const float4*)(Wa + o0);
      float4 a1 = *(const float4*)(Wa + o1);
      float4 b0 = *(const float4*)(Wb + o0);
      float4 b1v = *(const float4*)(Wb + o1);
      lds_wa32[(c4 + 0) * 16 + kk2] = pack_bf16x2(a0.x, a1.x);
      lds_wa32[(c4 + 1) * 16 + kk2] = pack_bf16x2(a0.y, a1.y);
      lds_wa32[(c4 + 2) * 16 + kk2] = pack_bf16x2(a0.z, a1.z);
      lds_wa32[(c4 + 3) * 16 + kk2] = pack_bf16x2(a0.w, a1.w);
      lds_wb32[(c4 + 0) * 16 + kk2] = pack_bf16x2(b0.x, b1v.x);
      lds_wb32[(c4 + 1) * 16 + kk2] = pack_bf16x2(b0.y, b1v.y);
      lds_wb32[(c4 + 2) * 16 + kk2] = pack_bf16x2(b0.z, b1v.z);
      lds_wb32[(c4 + 3) * 16 + kk2] = pack_bf16x2(b0.w, b1v.w);
    }
    __syncthreads();

    const bf16_t* arow = &lds_x[(mtile * 16 + ln) * XSTR + kc * 32];
    v16bf af = frag_pair(arow + 16 * half, arow + 16 * half + 8);
#pragma unroll
    for (int ntl = 0; ntl < 8; ++ntl) {
      int d = (nb2 + ntl) * 16 + ln;
      const bf16_t* pa = lds_wa + d * 32 + 16 * half;
      const bf16_t* pb = lds_wb + d * 32 + 16 * half;
      v16bf bfa = frag_pair(pa, pa + 8);
      v16bf bfb = frag_pair(pb, pb + 8);
      acca[ntl] = wmma_bf16(af, bfa, acca[ntl]);
      accg[ntl] = wmma_bf16(af, bfb, accg[ntl]);
    }
    __syncthreads();
  }

  // tanh * sigmoid, dot with Wc, reduce over D
  float sp[8] = {};
#pragma unroll
  for (int ntl = 0; ntl < 8; ++ntl) {
    int d = (nb2 + ntl) * 16 + ln;
    float biasa = ba[d], biasb = bb[d], wc = Wc[d];
#pragma unroll
    for (int r = 0; r < 8; ++r) {
      float a = tanhf(acca[ntl][r] + biasa);
      float g = 1.0f / (1.0f + __expf(-(accg[ntl][r] + biasb)));
      sp[r] += a * g * wc;
    }
  }
#pragma unroll
  for (int r = 0; r < 8; ++r) {
    float v = sp[r];
    v += __shfl_xor(v, 1, 32);
    v += __shfl_xor(v, 2, 32);
    v += __shfl_xor(v, 4, 32);
    v += __shfl_xor(v, 8, 32);
    sp[r] = v;
  }
  if (ln == 0) {
#pragma unroll
    for (int r = 0; r < 8; ++r)
      atomicAdd(&lds_s[mtile * 16 + half * 8 + r], sp[r]);
  }
  __syncthreads();

  if (t < TROWS) {
    int gr = row0 + t;
    if (gr < NROWS) {
      float s = lds_s[t] + bc[0];
      a_raw[gr]   = s;
      scoresA[gr] = s;
      scoresB[gr] = s;
    }
  }
}

// ---------------------------------------------------------------------------
// Kernel 2: global max of scores + zero accumulators (fresh every launch)
// ---------------------------------------------------------------------------
__global__ void k_reduce_max(const float* __restrict__ scores,
                             float* __restrict__ maxv, float* __restrict__ Z,
                             float* __restrict__ Mvec) {
  __shared__ float red[256];
  int t = threadIdx.x;
  float m = -__builtin_inff();
  for (int i = t; i < NROWS; i += 256) m = fmaxf(m, scores[i]);
  red[t] = m; __syncthreads();
  for (int s = 128; s > 0; s >>= 1) {
    if (t < s) red[t] = fmaxf(red[t], red[t + s]);
    __syncthreads();
  }
  if (t == 0) { *maxv = red[0]; *Z = 0.f; }
  for (int i = t; i < HDIM; i += 256) Mvec[i] = 0.f;
}

// ---------------------------------------------------------------------------
// Kernel 3: M_un = sum_n exp(s_n - max) * x_n ; Z = sum exp
// ---------------------------------------------------------------------------
__global__ void __launch_bounds__(256)
k_pool(const bf16_t* __restrict__ xbf, const float* __restrict__ scores,
       const float* __restrict__ maxv, float* __restrict__ Z,
       float* __restrict__ Mvec) {
  __shared__ float lm[HDIM];
  __shared__ float lz;
  int t = threadIdx.x;
  if (t == 0) lz = 0.f;
  for (int i = t; i < HDIM; i += 256) lm[i] = 0.f;
  __syncthreads();

  float mx = *maxv;
  int c8   = (t & 63) * 8;       // 8 cols per thread (one b128)
  int rsub = t >> 6;             // 4 concurrent rows per block
  float macc[8] = {};
  float zacc = 0.f;
  for (int n = blockIdx.x * 4 + rsub; n < NROWS; n += gridDim.x * 4) {
    float w = __expf(scores[n] - mx);
    if (c8 == 0) zacc += w;
    v8bf xv = *(const v8bf*)(xbf + (size_t)n * HDIM + c8);
#pragma unroll
    for (int j = 0; j < 8; ++j) macc[j] += w * (float)xv[j];
  }
#pragma unroll
  for (int j = 0; j < 8; ++j) atomicAdd(&lm[c8 + j], macc[j]);
  if (c8 == 0) atomicAdd(&lz, zacc);
  __syncthreads();
  for (int i = t; i < HDIM; i += 256) atomicAdd(&Mvec[i], lm[i]);
  if (t == 0) atomicAdd(Z, lz);
}

// ---------------------------------------------------------------------------
// Kernel 4: top-8 / bottom-8 indices (iterative arg-extremum, single block)
// ---------------------------------------------------------------------------
__global__ void k_topk(float* __restrict__ sA, float* __restrict__ sB,
                       int* __restrict__ topidx) {
  __shared__ float rv[256];
  __shared__ int   ri[256];
  int t = threadIdx.x;
  for (int it = 0; it < KSAMP; ++it) {
    float bv = -__builtin_inff(); int bi = 0;
    for (int i = t; i < NROWS; i += 256) {
      float v = sA[i];
      if (v > bv) { bv = v; bi = i; }
    }
    rv[t] = bv; ri[t] = bi; __syncthreads();
    for (int s = 128; s > 0; s >>= 1) {
      if (t < s && rv[t + s] > rv[t]) { rv[t] = rv[t + s]; ri[t] = ri[t + s]; }
      __syncthreads();
    }
    if (t == 0) { topidx[it] = ri[0]; sA[ri[0]] = -__builtin_inff(); }
    __syncthreads();
  }
  for (int it = 0; it < KSAMP; ++it) {
    float bv = __builtin_inff(); int bi = 0;
    for (int i = t; i < NROWS; i += 256) {
      float v = sB[i];
      if (v < bv) { bv = v; bi = i; }
    }
    rv[t] = bv; ri[t] = bi; __syncthreads();
    for (int s = 128; s > 0; s >>= 1) {
      if (t < s && rv[t + s] < rv[t]) { rv[t] = rv[t + s]; ri[t] = ri[t + s]; }
      __syncthreads();
    }
    if (t == 0) { topidx[KSAMP + it] = ri[0]; sB[ri[0]] = __builtin_inff(); }
    __syncthreads();
  }
}

// ---------------------------------------------------------------------------
// Kernel 5: bag logits / softmax / argmax + instance logits
// d_out layout: logits[2], Y_prob[2], Y_hat[1], A_raw[N], inst_logits[32]
// ---------------------------------------------------------------------------
__global__ void k_final(const float* __restrict__ Mvec, const float* __restrict__ Z,
                        const bf16_t* __restrict__ xbf, const int* __restrict__ topidx,
                        const float* __restrict__ Wcls, const float* __restrict__ bcls,
                        const float* __restrict__ Winst, const float* __restrict__ binst,
                        float* __restrict__ out) {
  __shared__ float red0[64], red1[64];
  __shared__ float Msh[HDIM];
  int t = threadIdx.x;  // 64 threads
  float z = *Z;
  for (int i = t; i < HDIM; i += 64) Msh[i] = Mvec[i] / z;
  __syncthreads();

  float p0 = 0.f, p1 = 0.f;
  for (int c = t; c < HDIM; c += 64) {
    p0 += Msh[c] * Wcls[c * 2 + 0];
    p1 += Msh[c] * Wcls[c * 2 + 1];
  }
  red0[t] = p0; red1[t] = p1; __syncthreads();
  for (int s = 32; s > 0; s >>= 1) {
    if (t < s) { red0[t] += red0[t + s]; red1[t] += red1[t + s]; }
    __syncthreads();
  }
  if (t == 0) {
    float l0 = red0[0] + bcls[0], l1 = red1[0] + bcls[1];
    out[0] = l0; out[1] = l1;
    float mx = fmaxf(l0, l1);
    float e0 = __expf(l0 - mx), e1 = __expf(l1 - mx);
    float inv = 1.f / (e0 + e1);
    out[2] = e0 * inv; out[3] = e1 * inv;
    out[4] = (l1 > l0) ? 1.0f : 0.0f;
  }
  __syncthreads();

  for (int inst = 0; inst < 2 * KSAMP; ++inst) {
    int idx = topidx[inst];
    float q0 = 0.f, q1 = 0.f;
    for (int c = t; c < HDIM; c += 64) {
      float xv = (float)xbf[(size_t)idx * HDIM + c];
      q0 += xv * Winst[c * 2 + 0];
      q1 += xv * Winst[c * 2 + 1];
    }
    red0[t] = q0; red1[t] = q1; __syncthreads();
    for (int s = 32; s > 0; s >>= 1) {
      if (t < s) { red0[t] += red0[t + s]; red1[t] += red1[t + s]; }
      __syncthreads();
    }
    if (t == 0) {
      out[5 + NROWS + inst * 2 + 0] = red0[0] + binst[0];
      out[5 + NROWS + inst * 2 + 1] = red1[0] + binst[1];
    }
    __syncthreads();
  }
}

// ---------------------------------------------------------------------------
extern "C" void kernel_launch(void* const* d_in, const int* in_sizes, int n_in,
                              void* d_out, int out_size, void* d_ws, size_t ws_size,
                              hipStream_t stream) {
  (void)in_sizes; (void)n_in; (void)out_size; (void)ws_size;
  const float* h     = (const float*)d_in[0];
  const float* W1    = (const float*)d_in[1];
  const float* b1    = (const float*)d_in[2];
  const float* Wa    = (const float*)d_in[3];
  const float* ba    = (const float*)d_in[4];
  const float* Wb    = (const float*)d_in[5];
  const float* bb    = (const float*)d_in[6];
  const float* Wc    = (const float*)d_in[7];
  const float* bc    = (const float*)d_in[8];
  const float* Wcls  = (const float*)d_in[9];
  const float* bcls  = (const float*)d_in[10];
  const float* Winst = (const float*)d_in[11];
  const float* binst = (const float*)d_in[12];
  float* out = (float*)d_out;

  char* ws = (char*)d_ws;
  size_t off = 0;
  bf16_t* xbf = (bf16_t*)(ws + off); off += (size_t)NROWS * HDIM * sizeof(bf16_t);
  off = (off + 255) & ~(size_t)255;
  float* scoresA = (float*)(ws + off); off += (size_t)NROWS * sizeof(float);
  float* scoresB = (float*)(ws + off); off += (size_t)NROWS * sizeof(float);
  float* maxv    = (float*)(ws + off); off += 256;
  float* Z       = (float*)(ws + off); off += 256;
  float* Mvec    = (float*)(ws + off); off += HDIM * sizeof(float);
  int*   topidx  = (int*)(ws + off);   off += 16 * sizeof(int);

  float* a_raw = out + 5;   // A_raw region of d_out

  int nblocks = (NROWS + TROWS - 1) / TROWS;   // 1563
  k_fused<<<nblocks, 256, 0, stream>>>(h, W1, b1, Wa, ba, Wb, bb, Wc, bc,
                                       xbf, a_raw, scoresA, scoresB);
  k_reduce_max<<<1, 256, 0, stream>>>(a_raw, maxv, Z, Mvec);
  k_pool<<<256, 256, 0, stream>>>(xbf, a_raw, maxv, Z, Mvec);
  k_topk<<<1, 256, 0, stream>>>(scoresA, scoresB, topidx);
  k_final<<<1, 64, 0, stream>>>(Mvec, Z, xbf, topidx, Wcls, bcls, Winst, binst, out);
}
